// Transformer_7499012899637
// MI455X (gfx1250) — matrix operand order repaired and checked by reading, not compiled, not run on hardware
//
#include <hip/hip_runtime.h>
#include <hip/hip_bf16.h>

typedef __attribute__((ext_vector_type(16))) _Float16 v16h;
typedef __attribute__((ext_vector_type(8)))  _Float16 v8h;
typedef __attribute__((ext_vector_type(8)))  float    v8f;

#define NTOK 3072
#define DMODEL 32
#define NHEAD 8
#define EDIM 256
#define NLAYER 2
#define QTILES 4   // query tiles (of 16) per attention wave

// ---------------- WMMA helpers ----------------

__device__ __forceinline__ v8f wmma16(v16h a, v16h b, v8f c) {
  return __builtin_amdgcn_wmma_f32_16x16x32_f16(
      /*neg_a=*/false, a, /*neg_b=*/false, b,
      /*c_mod=*/(short)0, c, /*reuse_a=*/false, /*reuse_b=*/false);
}

// A-operand (16x32, f16, row-major source). Lane = row (lane&15).
// lanes 0-15 hold K 0..15; lanes 16-31 hold K 16..31.
__device__ __forceinline__ v16h load_a32(const _Float16* __restrict__ base, int stride) {
  const int lane = threadIdx.x & 31;
  const int hi = lane >> 4;
  const _Float16* p = base + (lane & 15) * stride + hi * 16;
  v8h c0 = *reinterpret_cast<const v8h*>(p);
  v8h c1 = *reinterpret_cast<const v8h*>(p + 8);
  v16h a;
#pragma unroll
  for (int i = 0; i < 8; ++i) { a[i] = c0[i]; a[i + 8] = c1[i]; }
  return a;
}

// B-operand (32x16, f16) loaded from B^T stored row-major (row = output col).
// Lane = col (lane&15); lanes 0-15 hold K 0..15, lanes 16-31 hold K 16..31.
__device__ __forceinline__ v16h load_b32(const _Float16* __restrict__ base, int stride) {
  const int lane = threadIdx.x & 31;
  const int hi = lane >> 4;
  const _Float16* p = base + (lane & 15) * stride + hi * 16;
  v8h c0 = *reinterpret_cast<const v8h*>(p);
  v8h c1 = *reinterpret_cast<const v8h*>(p + 8);
  v16h b;
#pragma unroll
  for (int i = 0; i < 8; ++i) { b[i] = c0[i]; b[i + 8] = c1[i]; }
  return b;
}

// ---------------- weight prep: f32 -> f16 (+transpose) ----------------
__global__ void prep_weights_kernel(const float* __restrict__ wq,
                                    const float* __restrict__ wk,
                                    const float* __restrict__ wv,
                                    const float* __restrict__ wo,
                                    _Float16* __restrict__ wqt,
                                    _Float16* __restrict__ wkt,
                                    _Float16* __restrict__ wvt,
                                    _Float16* __restrict__ wot) {
  int tid = blockIdx.x * blockDim.x + threadIdx.x;
  if (tid >= NLAYER * 4 * 8192) return;
  int l = tid >> 15;
  int r = tid & 32767;
  int which = r >> 13;
  int e = r & 8191;
  if (which < 3) {
    const float* src = (which == 0) ? wq : (which == 1) ? wk : wv;
    _Float16* dst = (which == 0) ? wqt : (which == 1) ? wkt : wvt;
    int c = e >> 5;
    int kk = e & 31;
    dst[l * 8192 + c * 32 + kk] = (_Float16)src[l * 8192 + kk * 256 + c];
  } else {
    int c = e >> 8;
    int kk = e & 255;
    wot[l * 8192 + c * 256 + kk] = (_Float16)wo[l * 8192 + kk * 32 + c];
  }
}

// ---------------- encoder: x[N,6] -> h[N,32] (f32, in d_out) ----------------
__global__ void encoder_kernel(const float* __restrict__ x,
                               const float* __restrict__ w1, const float* __restrict__ b1,
                               const float* __restrict__ w2, const float* __restrict__ b2,
                               float* __restrict__ h) {
  int t = blockIdx.x * blockDim.x + threadIdx.x;
  if (t >= NTOK) return;
  float xi[6];
#pragma unroll
  for (int i = 0; i < 6; ++i) xi[i] = x[t * 6 + i];
  float hid[32];
#pragma unroll
  for (int j = 0; j < 32; ++j) {
    float s = b1[j];
#pragma unroll
    for (int i = 0; i < 6; ++i) s += xi[i] * w1[i * 32 + j];
    hid[j] = fmaxf(s, 0.0f);
  }
  for (int j = 0; j < 32; ++j) {
    float s = b2[j];
#pragma unroll
    for (int i = 0; i < 32; ++i) s += hid[i] * w2[i * 32 + j];
    h[t * 32 + j] = s;
  }
}

// ---------------- LN1: h[N,32] f32 -> xn[N,32] f16 ----------------
__global__ void ln_kernel(const float* __restrict__ h,
                          const float* __restrict__ g, const float* __restrict__ b,
                          _Float16* __restrict__ xn) {
  int t = blockIdx.x * blockDim.x + threadIdx.x;
  if (t >= NTOK) return;
  float buf[32];
  float mean = 0.f;
#pragma unroll
  for (int i = 0; i < 32; ++i) { buf[i] = h[t * 32 + i]; mean += buf[i]; }
  mean *= (1.0f / 32.0f);
  float var = 0.f;
#pragma unroll
  for (int i = 0; i < 32; ++i) { float d = buf[i] - mean; var += d * d; }
  var *= (1.0f / 32.0f);
  float rs = rsqrtf(var + 1e-5f);
#pragma unroll
  for (int i = 0; i < 32; ++i)
    xn[t * 32 + i] = (_Float16)((buf[i] - mean) * rs * g[i] + b[i]);
}

// ---------------- QKV projection via WMMA ----------------
// grid: (192 token tiles, 48 col tiles). y<16: Q (pre-scaled), y<32: K, else V^T.
__global__ void __launch_bounds__(32)
qkv_kernel(const _Float16* __restrict__ xn,
           const _Float16* __restrict__ wqt, const _Float16* __restrict__ wkt,
           const _Float16* __restrict__ wvt,
           _Float16* __restrict__ q, _Float16* __restrict__ k,
           _Float16* __restrict__ vt) {
  const int tile = blockIdx.x;
  const int y = blockIdx.y;
  const int sel = y >> 4;
  const int ct = y & 15;
  const _Float16* wt = (sel == 0) ? wqt : (sel == 1) ? wkt : wvt;

  v16h a = load_a32(xn + tile * 16 * 32, 32);
  v16h b = load_b32(wt + ct * 16 * 32, 32);
  v8f z = {};
  v8f d = wmma16(a, b, z);

  const int lane = threadIdx.x & 31;
  const int n = lane & 15;
  const int hi = lane >> 4;
  const int col = ct * 16 + n;
  if (sel < 2) {
    const float sc = (sel == 0) ? 0.1767766952966369f : 1.0f;  // fold 1/sqrt(D) into Q
    _Float16* out = (sel == 0) ? q : k;
#pragma unroll
    for (int r = 0; r < 8; ++r)
      out[(tile * 16 + hi * 8 + r) * EDIM + col] = (_Float16)(d[r] * sc);
  } else {
    v8h pk;
#pragma unroll
    for (int r = 0; r < 8; ++r) pk[r] = (_Float16)d[r];
    *reinterpret_cast<v8h*>(vt + col * NTOK + tile * 16 + hi * 8) = pk;
  }
}

// ---------------- flash attention, one wave = (head, 4 x 16-query tiles) ----------------
__global__ void __launch_bounds__(32)
attn_kernel(const _Float16* __restrict__ q, const _Float16* __restrict__ k,
            const _Float16* __restrict__ vt, _Float16* __restrict__ o) {
  const int qg = blockIdx.x;   // 0..47 (group of QTILES query tiles)
  const int hh = blockIdx.y;   // 0..7
  const int lane = threadIdx.x & 31;
  const int hi = lane >> 4;

  const _Float16* kh  = k + hh * 32;
  const _Float16* vth = vt + hh * 32 * NTOK;

  // B = Q^T fragments (Q already scaled by 1/sqrt(D))
  v16h bq[QTILES];
#pragma unroll
  for (int t = 0; t < QTILES; ++t)
    bq[t] = load_b32(q + (qg * QTILES + t) * 16 * EDIM + hh * 32, EDIM);

  float m[QTILES], l[QTILES];
  v8f of0[QTILES], of1[QTILES];
#pragma unroll
  for (int t = 0; t < QTILES; ++t) {
    m[t] = -1e30f; l[t] = 0.f;
    of0[t] = (v8f){}; of1[t] = (v8f){};
  }

  for (int jb = 0; jb < NTOK; jb += 32) {
    // shared per-wave loads: K rows and V^T rows for this 32-key block
    v16h a0  = load_a32(kh + jb * EDIM, EDIM);
    v16h a1  = load_a32(kh + (jb + 16) * EDIM, EDIM);
    v16h av0 = load_a32(vth + jb, NTOK);
    v16h av1 = load_a32(vth + 16 * NTOK + jb, NTOK);

    if (jb + 128 < NTOK) {  // stream-ahead hints (global_prefetch_b8)
      __builtin_prefetch(kh + (jb + 128) * EDIM, 0, 0);
      __builtin_prefetch(vth + jb + 128, 0, 0);
    }

#pragma unroll
    for (int t = 0; t < QTILES; ++t) {
      v8f z = {};
      v8f s0 = wmma16(a0, bq[t], z);   // S^T keys jb+{0..15}
      v8f s1 = wmma16(a1, bq[t], z);   // S^T keys jb+{16..31}

      float tm = -1e30f;
#pragma unroll
      for (int r = 0; r < 8; ++r) tm = fmaxf(tm, fmaxf(s0[r], s1[r]));
      tm = fmaxf(tm, __shfl_xor(tm, 16, 32));
      float mn = fmaxf(m[t], tm);
      float alpha = __expf(m[t] - mn);

      v8f p0, p1;
      float rs = 0.f;
#pragma unroll
      for (int r = 0; r < 8; ++r) {
        p0[r] = __expf(s0[r] - mn);
        p1[r] = __expf(s1[r] - mn);
        rs += p0[r] + p1[r];
      }
      rs += __shfl_xor(rs, 16, 32);
      l[t] = l[t] * alpha + rs;
      m[t] = mn;
#pragma unroll
      for (int r = 0; r < 8; ++r) { of0[t][r] *= alpha; of1[t][r] *= alpha; }

      // pack P to f16 pairs, then exchange lane halves (8 dword shuffles)
      union H2 { _Float16 h[2]; unsigned u; };
      unsigned pk0[4], pk1[4];
#pragma unroll
      for (int i = 0; i < 4; ++i) {
        H2 u0; u0.h[0] = (_Float16)p0[2 * i]; u0.h[1] = (_Float16)p0[2 * i + 1];
        H2 u1; u1.h[0] = (_Float16)p1[2 * i]; u1.h[1] = (_Float16)p1[2 * i + 1];
        pk0[i] = u0.u; pk1[i] = u1.u;
      }
      unsigned x0[4], x1[4];
#pragma unroll
      for (int i = 0; i < 4; ++i) {
        x0[i] = (unsigned)__shfl_xor((int)pk0[i], 16, 32);
        x1[i] = (unsigned)__shfl_xor((int)pk1[i], 16, 32);
      }
      union BP { v16h v; unsigned u[8]; } bp;
#pragma unroll
      for (int i = 0; i < 4; ++i) {
        bp.u[i]     = hi ? x1[i] : pk0[i];   // keys 16*hi + {0..7}
        bp.u[4 + i] = hi ? pk1[i] : x0[i];   // keys 16*hi + {8..15}
      }

      of0[t] = wmma16(av0, bp.v, of0[t]);
      of1[t] = wmma16(av1, bp.v, of1[t]);
    }
  }

#pragma unroll
  for (int t = 0; t < QTILES; ++t) {
    float inv = 1.0f / l[t];
    int tok = (qg * QTILES + t) * 16 + (lane & 15);
    v8h o0, o1;
#pragma unroll
    for (int r = 0; r < 8; ++r) {
      o0[r] = (_Float16)(of0[t][r] * inv);
      o1[r] = (_Float16)(of1[t][r] * inv);
    }
    *reinterpret_cast<v8h*>(o + tok * EDIM + hh * 32 + hi * 8) = o0;
    *reinterpret_cast<v8h*>(o + tok * EDIM + hh * 32 + 16 + hi * 8) = o1;
  }
}

// ---------------- O projection + residual: h += o @ wo + bo ----------------
__global__ void __launch_bounds__(32)
oproj_kernel(const _Float16* __restrict__ o, const _Float16* __restrict__ wot,
             const float* __restrict__ bo, float* __restrict__ h) {
  const int tile = blockIdx.x;
  const int nt = blockIdx.y;
  v8f acc = {};
#pragma unroll
  for (int kk = 0; kk < 8; ++kk) {
    v16h a = load_a32(o + tile * 16 * EDIM + kk * 32, EDIM);
    v16h b = load_b32(wot + nt * 16 * EDIM + kk * 32, EDIM);
    acc = wmma16(a, b, acc);
  }
  const int lane = threadIdx.x & 31;
  const int n = lane & 15;
  const int hi = lane >> 4;
  const int col = nt * 16 + n;
  float bias = bo[col];
#pragma unroll
  for (int r = 0; r < 8; ++r) {
    int tok = tile * 16 + hi * 8 + r;
    h[tok * 32 + col] += acc[r] + bias;
  }
}

// ---------------- LN2 + FFN + residual (thread per token) ----------------
__global__ void ff_kernel(float* __restrict__ h,
                          const float* __restrict__ g, const float* __restrict__ b,
                          const float* __restrict__ w1, const float* __restrict__ b1,
                          const float* __restrict__ w2, const float* __restrict__ b2) {
  int t = blockIdx.x * blockDim.x + threadIdx.x;
  if (t >= NTOK) return;
  float buf[32];
  float mean = 0.f;
#pragma unroll
  for (int i = 0; i < 32; ++i) { buf[i] = h[t * 32 + i]; mean += buf[i]; }
  mean *= (1.0f / 32.0f);
  float var = 0.f;
#pragma unroll
  for (int i = 0; i < 32; ++i) { float d = buf[i] - mean; var += d * d; }
  var *= (1.0f / 32.0f);
  float rs = rsqrtf(var + 1e-5f);
  float xn[32];
#pragma unroll
  for (int i = 0; i < 32; ++i) xn[i] = (buf[i] - mean) * rs * g[i] + b[i];
  float f[32];
  for (int j = 0; j < 32; ++j) {
    float s = b1[j];
#pragma unroll
    for (int i = 0; i < 32; ++i) s += xn[i] * w1[i * 32 + j];
    f[j] = fmaxf(s, 0.0f);
  }
  for (int j = 0; j < 32; ++j) {
    float s = b2[j];
#pragma unroll
    for (int i = 0; i < 32; ++i) s += f[i] * w2[i * 32 + j];
    h[t * 32 + j] = buf[j] + s;
  }
}

// ---------------- host launcher ----------------
extern "C" void kernel_launch(void* const* d_in, const int* in_sizes, int n_in,
                              void* d_out, int out_size, void* d_ws, size_t ws_size,
                              hipStream_t stream) {
  const float* x      = (const float*)d_in[0];
  const float* enc_w1 = (const float*)d_in[1];
  const float* enc_b1 = (const float*)d_in[2];
  const float* enc_w2 = (const float*)d_in[3];
  const float* enc_b2 = (const float*)d_in[4];
  const float* ln1_g  = (const float*)d_in[5];
  const float* ln1_b  = (const float*)d_in[6];
  const float* wq     = (const float*)d_in[7];
  const float* wk     = (const float*)d_in[8];
  const float* wv     = (const float*)d_in[9];
  const float* wo     = (const float*)d_in[10];
  const float* bo     = (const float*)d_in[11];
  const float* ln2_g  = (const float*)d_in[12];
  const float* ln2_b  = (const float*)d_in[13];
  const float* ff1_w  = (const float*)d_in[14];
  const float* ff1_b  = (const float*)d_in[15];
  const float* ff2_w  = (const float*)d_in[16];
  const float* ff2_b  = (const float*)d_in[17];

  float* h = (float*)d_out;   // running residual stream, [3072,32] f32

  size_t off = 0;
  auto carve = [&](size_t halfs) -> _Float16* {
    _Float16* p = (_Float16*)((char*)d_ws + off);
    off += ((halfs * sizeof(_Float16) + 255) / 256) * 256;
    return p;
  };
  _Float16* xn  = carve((size_t)NTOK * DMODEL);
  _Float16* qb  = carve((size_t)NTOK * EDIM);
  _Float16* kb  = carve((size_t)NTOK * EDIM);
  _Float16* vtb = carve((size_t)NTOK * EDIM);   // transposed: [E][N]
  _Float16* ob  = carve((size_t)NTOK * EDIM);
  _Float16* wqt = carve((size_t)NLAYER * 8192);
  _Float16* wkt = carve((size_t)NLAYER * 8192);
  _Float16* wvt = carve((size_t)NLAYER * 8192);
  _Float16* wot = carve((size_t)NLAYER * 8192);

  prep_weights_kernel<<<dim3(256), dim3(256), 0, stream>>>(
      wq, wk, wv, wo, wqt, wkt, wvt, wot);

  encoder_kernel<<<dim3(12), dim3(256), 0, stream>>>(
      x, enc_w1, enc_b1, enc_w2, enc_b2, h);

  for (int l = 0; l < NLAYER; ++l) {
    ln_kernel<<<dim3(12), dim3(256), 0, stream>>>(
        h, ln1_g + l * 32, ln1_b + l * 32, xn);

    qkv_kernel<<<dim3(NTOK / 16, 48), dim3(32), 0, stream>>>(
        xn, wqt + l * 8192, wkt + l * 8192, wvt + l * 8192, qb, kb, vtb);

    attn_kernel<<<dim3(NTOK / (16 * QTILES), NHEAD), dim3(32), 0, stream>>>(
        qb, kb, vtb, ob);

    oproj_kernel<<<dim3(NTOK / 16, 2), dim3(32), 0, stream>>>(
        ob, wot + l * 8192, bo + l * 32, h);

    ff_kernel<<<dim3(12), dim3(256), 0, stream>>>(
        h, ln2_g + l * 32, ln2_b + l * 32,
        ff1_w + l * 1024, ff1_b + l * 32,
        ff2_w + l * 1024, ff2_b + l * 32);
  }
}
